// RiskGNN_68942815036075
// MI455X (gfx1250) — matrix operand order repaired and checked by reading, not compiled, not run on hardware
//
#include <hip/hip_runtime.h>
#include <hip/hip_bf16.h>
#include <math.h>

// ---------------- problem constants ----------------
static constexpr int NCOMP = 100000;
static constexpr int NPERS = 50000;
static constexpr int NN    = 150000;   // NCOMP + NPERS
static constexpr int DD    = 128;
static constexpr int RR    = 7;
static constexpr int EPR   = 300000;   // edges per relation
static constexpr float NEG_SLOPE = 0.2f;

// ---------------- WMMA types / helpers ----------------
typedef __attribute__((ext_vector_type(16))) __bf16 v16bf;
typedef __attribute__((ext_vector_type(8)))  __bf16 v8bf;
typedef __attribute__((ext_vector_type(8)))  float  v8f;

__device__ __forceinline__ v8f wmma_bf(const v16bf& a, const v16bf& b, const v8f& c) {
  return __builtin_amdgcn_wmma_f32_16x16x32_bf16(false, a, false, b, (short)0, c, false, false);
}

// A fragment, 16x32 bf16 (ISA 7.12.2): lane m=lane&15 holds K = k0+8h..+7 and k0+16+8h..+7
__device__ __forceinline__ v16bf ld_a_bf(const __bf16* p0, int h) {
  const __bf16* p = p0 + 8 * h;
  v8bf lo = *(const v8bf*)p;
  v8bf hi = *(const v8bf*)(p + 16);
  return __builtin_shufflevector(lo, hi, 0,1,2,3,4,5,6,7,8,9,10,11,12,13,14,15);
}
__device__ __forceinline__ v16bf ld_a_f32(const float* row, int k0, int h) {
  const float* p = row + k0 + 8 * h;
  v16bf r;
#pragma unroll
  for (int j = 0; j < 8; ++j) { r[j] = (__bf16)p[j]; r[8 + j] = (__bf16)p[16 + j]; }
  return r;
}
__device__ __forceinline__ v16bf ld_a_f32_lrelu(const float* row, int k0, int h) {
  const float* p = row + k0 + 8 * h;
  v16bf r;
#pragma unroll
  for (int j = 0; j < 8; ++j) {
    float lo = p[j], hi = p[16 + j];
    lo = lo > 0.f ? lo : NEG_SLOPE * lo;
    hi = hi > 0.f ? hi : NEG_SLOPE * hi;
    r[j] = (__bf16)lo; r[8 + j] = (__bf16)hi;
  }
  return r;
}
// B fragment from column-major LDS weights WT[n*KD + k]: lane half h holds K = k0+8h..+7 and k0+16+8h..+7
__device__ __forceinline__ v16bf ld_b(const __bf16* WT, int KD, int ncol, int k0, int h) {
  const __bf16* p = WT + (size_t)ncol * KD + k0 + 8 * h;
  v8bf lo = *(const v8bf*)p;
  v8bf hi = *(const v8bf*)(p + 16);
  return __builtin_shufflevector(lo, hi, 0,1,2,3,4,5,6,7,8,9,10,11,12,13,14,15);
}

// monotonic float<->uint encoding for atomic max
__device__ __forceinline__ unsigned enc_f32(float f) {
  unsigned u = __float_as_uint(f);
  return (u & 0x80000000u) ? ~u : (u | 0x80000000u);
}
__device__ __forceinline__ float dec_f32(unsigned u) {
  return (u & 0x80000000u) ? __uint_as_float(u & 0x7FFFFFFFu) : __uint_as_float(~u);
}
static constexpr unsigned ENC_NEG_INF = 0x007FFFFFu; // enc(-inf)

// ---------------- init kernels ----------------
__global__ void k_zero(float* p, long long n) {
  long long i = (long long)blockIdx.x * blockDim.x + threadIdx.x;
  long long st = (long long)gridDim.x * blockDim.x;
  for (; i < n; i += st) p[i] = 0.f;
}
__global__ void k_init_ms(unsigned* m, float* s, long long n) {
  long long i = (long long)blockIdx.x * blockDim.x + threadIdx.x;
  long long st = (long long)gridDim.x * blockDim.x;
  for (; i < n; i += st) { m[i] = ENC_NEG_INF; s[i] = 0.f; }
}

// ---------------- node embedding GEMM ----------------
__global__ void __launch_bounds__(128) k_embed(const float* __restrict__ com,
    const float* __restrict__ per, const float* __restrict__ Wc,
    const float* __restrict__ Wp, float* __restrict__ emb, __bf16* __restrict__ embh) {
  __shared__ __align__(32) __bf16 WTc[DD * DD];
  __shared__ __align__(32) __bf16 WTp[DD * DD];
  for (int idx = threadIdx.x; idx < DD * DD; idx += 128) {
    int k = idx >> 7, n = idx & 127;
    WTc[n * DD + k] = (__bf16)Wc[idx];
    WTp[n * DD + k] = (__bf16)Wp[idx];
  }
  __syncthreads();
  const int wave = threadIdx.x >> 5, lane = threadIdx.x & 31;
  const int h = lane >> 4, nl = lane & 15;
  const int row0 = blockIdx.x * 64 + wave * 16;
  if (row0 >= NN) return;
  const int arow = row0 + nl;
  const float* abase = (arow < NCOMP) ? (com + (size_t)arow * DD)
                                      : (per + (size_t)(arow - NCOMP) * DD);
  const __bf16* WT = (row0 < NCOMP) ? WTc : WTp;   // NCOMP%16==0 -> uniform per wave
  v16bf a[4];
#pragma unroll
  for (int ks = 0; ks < 4; ++ks) a[ks] = ld_a_f32(abase, ks * 32, h);
#pragma unroll
  for (int nt = 0; nt < 8; ++nt) {
    v8f c = {};
#pragma unroll
    for (int ks = 0; ks < 4; ++ks) c = wmma_bf(a[ks], ld_b(WT, DD, nt * 16 + nl, ks * 32, h), c);
#pragma unroll
    for (int j = 0; j < 8; ++j) {
      size_t o = (size_t)(row0 + 8 * h + j) * DD + nt * 16 + nl;
      emb[o] = c[j];
      embh[o] = (__bf16)c[j];
    }
  }
}

// -------- attention relation pass 1: temp = [emb[dst],emb[src]] @ Wpair, seg-max --------
__global__ void __launch_bounds__(128) k_edge_attn(const __bf16* __restrict__ embh,
    const int* __restrict__ eidx, const float* __restrict__ Wpair,
    float* __restrict__ temp, unsigned* __restrict__ menc, int e0) {
  __shared__ __align__(32) __bf16 WT[2 * DD * DD]; // col-major [n][256]
  for (int idx = threadIdx.x; idx < 2 * DD * DD; idx += 128) {
    int k = idx >> 7, n = idx & 127;
    WT[n * (2 * DD) + k] = (__bf16)Wpair[idx];
  }
  __syncthreads();
  const int wave = threadIdx.x >> 5, lane = threadIdx.x & 31;
  const int h = lane >> 4, nl = lane & 15;
  const int t0 = blockIdx.x * 64 + wave * 16;
  if (t0 >= EPR) return;
  const int e = e0 + t0 + nl;
  const int esrc = eidx[2 * e], edst = eidx[2 * e + 1];
  const __bf16* srow = embh + (size_t)esrc * DD;
  const __bf16* drow = embh + (size_t)edst * DD;
  v16bf a[8];
#pragma unroll
  for (int ks = 0; ks < 8; ++ks) {
    int k0 = ks * 32;
    a[ks] = (k0 < DD) ? ld_a_bf(drow + k0, h) : ld_a_bf(srow + (k0 - DD), h);
  }
  int dstm[8];
#pragma unroll
  for (int j = 0; j < 8; ++j) dstm[j] = __shfl(edst, 8 * h + j);
#pragma unroll
  for (int nt = 0; nt < 8; ++nt) {
    v8f c = {};
#pragma unroll
    for (int ks = 0; ks < 8; ++ks) c = wmma_bf(a[ks], ld_b(WT, 2 * DD, nt * 16 + nl, ks * 32, h), c);
#pragma unroll
    for (int j = 0; j < 8; ++j) {
      temp[(size_t)(t0 + 8 * h + j) * DD + nt * 16 + nl] = c[j];
      atomicMax(menc + (size_t)dstm[j] * DD + nt * 16 + nl, enc_f32(c[j]));
    }
  }
}

// -------- pass 2: e = exp(t - m[dst]); seg-sum --------
__global__ void k_exp_sum(const int* __restrict__ eidx, float* __restrict__ temp,
                          const unsigned* __restrict__ menc, float* __restrict__ s, int e0) {
  long long gid = (long long)blockIdx.x * 256 + threadIdx.x;
  if (gid >= (long long)EPR * DD) return;
  int er = (int)(gid >> 7), n = (int)(gid & 127);
  int dst = eidx[2 * (e0 + er) + 1];
  float mm = dec_f32(menc[(size_t)dst * DD + n]);
  if (!(mm > -3.0e38f)) mm = 0.0f;
  float ex = __expf(temp[gid] - mm);
  temp[gid] = ex;
  atomicAdd(s + (size_t)dst * DD + n, ex);
}

// -------- pass 3: msg = x_j * e/s ; scatter-add into rs_r --------
__global__ void k_msg(const int* __restrict__ eidx, const float* __restrict__ temp,
                      const float* __restrict__ s, const float* __restrict__ emb,
                      float* __restrict__ rsr, int e0) {
  long long gid = (long long)blockIdx.x * 256 + threadIdx.x;
  if (gid >= (long long)EPR * DD) return;
  int er = (int)(gid >> 7), n = (int)(gid & 127);
  int src = eidx[2 * (e0 + er)], dst = eidx[2 * (e0 + er) + 1];
  float w = temp[gid];
  float xj = emb[(size_t)src * DD + n];
  float den = s[(size_t)dst * DD + n] + 1e-16f;
  atomicAdd(rsr + (size_t)dst * DD + n, xj * w / den);
}

// -------- relation 3: scalar edge-weight softmax --------
__global__ void k_w_max3(const int* __restrict__ eidx, const float* __restrict__ ew,
                         unsigned* __restrict__ m3, int e0) {
  int t = blockIdx.x * 256 + threadIdx.x;
  if (t >= EPR) return;
  atomicMax(m3 + eidx[2 * (e0 + t) + 1], enc_f32(ew[e0 + t]));
}
__global__ void k_w_sum3(const int* __restrict__ eidx, const float* __restrict__ ew,
                         const unsigned* __restrict__ m3, float* __restrict__ s3, int e0) {
  int t = blockIdx.x * 256 + threadIdx.x;
  if (t >= EPR) return;
  int dst = eidx[2 * (e0 + t) + 1];
  float mm = dec_f32(m3[dst]);
  if (!(mm > -3.0e38f)) mm = 0.0f;
  atomicAdd(s3 + dst, __expf(ew[e0 + t] - mm));
}
__global__ void __launch_bounds__(128) k_edge3(const __bf16* __restrict__ embh,
    const int* __restrict__ eidx, const float* __restrict__ W3,
    const float* __restrict__ ew, const unsigned* __restrict__ m3,
    const float* __restrict__ s3, float* __restrict__ rs3, int e0) {
  __shared__ __align__(32) __bf16 WT[DD * DD];
  for (int idx = threadIdx.x; idx < DD * DD; idx += 128) {
    int k = idx >> 7, n = idx & 127;
    WT[n * DD + k] = (__bf16)W3[idx];
  }
  __syncthreads();
  const int wave = threadIdx.x >> 5, lane = threadIdx.x & 31;
  const int h = lane >> 4, nl = lane & 15;
  const int t0 = blockIdx.x * 64 + wave * 16;
  if (t0 >= EPR) return;
  const int e = e0 + t0 + nl;
  const int esrc = eidx[2 * e], edst = eidx[2 * e + 1];
  float mm = dec_f32(m3[edst]);
  if (!(mm > -3.0e38f)) mm = 0.0f;
  float coef = __expf(ew[e] - mm) / (s3[edst] + 1e-16f);
  const __bf16* srow = embh + (size_t)esrc * DD;
  v16bf a[4];
#pragma unroll
  for (int ks = 0; ks < 4; ++ks) a[ks] = ld_a_bf(srow + ks * 32, h);
  int dstm[8]; float cm[8];
#pragma unroll
  for (int j = 0; j < 8; ++j) { dstm[j] = __shfl(edst, 8 * h + j); cm[j] = __shfl(coef, 8 * h + j); }
#pragma unroll
  for (int nt = 0; nt < 8; ++nt) {
    v8f c = {};
#pragma unroll
    for (int ks = 0; ks < 4; ++ks) c = wmma_bf(a[ks], ld_b(WT, DD, nt * 16 + nl, ks * 32, h), c);
#pragma unroll
    for (int j = 0; j < 8; ++j)
      atomicAdd(rs3 + (size_t)dstm[j] * DD + nt * 16 + nl, c[j] * cm[j]);
  }
}

// -------- cross-relation scores: (emb@Wq_r+bq)·(LReLU(rs_r)@Wk_r+bk) --------
__global__ void __launch_bounds__(128) k_scores(const __bf16* __restrict__ embh,
    const float* __restrict__ rs, const float* __restrict__ Wq, const float* __restrict__ bq,
    const float* __restrict__ Wk, const float* __restrict__ bk,
    const float* __restrict__ pri, float* __restrict__ scores) {
  const int r = blockIdx.y;
  __shared__ __align__(32) __bf16 WTq[DD * DD];
  __shared__ __align__(32) __bf16 WTk[DD * DD];
  const float* Wqr = Wq + (size_t)r * DD * DD;
  const float* Wkr = Wk + (size_t)r * DD * DD;
  for (int idx = threadIdx.x; idx < DD * DD; idx += 128) {
    int k = idx >> 7, n = idx & 127;
    WTq[n * DD + k] = (__bf16)Wqr[idx];
    WTk[n * DD + k] = (__bf16)Wkr[idx];
  }
  __syncthreads();
  const int wave = threadIdx.x >> 5, lane = threadIdx.x & 31;
  const int h = lane >> 4, nl = lane & 15;
  const int row0 = blockIdx.x * 64 + wave * 16;
  if (row0 >= NN) return;
  const int arow = row0 + nl;
  const __bf16* eb = embh + (size_t)arow * DD;
  const float* rb = rs + (size_t)r * NN * DD + (size_t)arow * DD;
  v16bf aq[4], ak[4];
#pragma unroll
  for (int ks = 0; ks < 4; ++ks) {
    aq[ks] = ld_a_bf(eb + ks * 32, h);
    ak[ks] = ld_a_f32_lrelu(rb, ks * 32, h);
  }
  float part[8] = {0.f, 0.f, 0.f, 0.f, 0.f, 0.f, 0.f, 0.f};
#pragma unroll
  for (int nt = 0; nt < 8; ++nt) {
    v8f cq = {}, ck = {};
#pragma unroll
    for (int ks = 0; ks < 4; ++ks) {
      cq = wmma_bf(aq[ks], ld_b(WTq, DD, nt * 16 + nl, ks * 32, h), cq);
      ck = wmma_bf(ak[ks], ld_b(WTk, DD, nt * 16 + nl, ks * 32, h), ck);
    }
    float bqv = bq[r * DD + nt * 16 + nl];
    float bkv = bk[r * DD + nt * 16 + nl];
#pragma unroll
    for (int j = 0; j < 8; ++j) part[j] += (cq[j] + bqv) * (ck[j] + bkv);
  }
#pragma unroll
  for (int off = 1; off < 16; off <<= 1)
#pragma unroll
    for (int j = 0; j < 8; ++j) part[j] += __shfl_xor(part[j], off);
  if (nl == 0) {
    float sc = pri[r] * 0.08838834764831845f; // 1/sqrt(128)
#pragma unroll
    for (int j = 0; j < 8; ++j)
      scores[(size_t)r * NN + row0 + 8 * h + j] = part[j] * sc;
  }
}

// -------- output: softmax_r(scores) weighted v_r = LReLU(rs_r)@Wv+bv, + gelu skip --------
__global__ void __launch_bounds__(128) k_out(const float* __restrict__ rs,
    const float* __restrict__ Wv, const float* __restrict__ bv,
    const float* __restrict__ scores, const float* __restrict__ emb,
    const float* __restrict__ skip, float* __restrict__ out) {
  __shared__ __align__(32) __bf16 WT[DD * DD];
  __shared__ float att[4 * RR * 16];
  for (int idx = threadIdx.x; idx < DD * DD; idx += 128) {
    int k = idx >> 7, n = idx & 127;
    WT[n * DD + k] = (__bf16)Wv[idx];
  }
  const int wave = threadIdx.x >> 5, lane = threadIdx.x & 31;
  const int h = lane >> 4, nl = lane & 15;
  const int row0 = blockIdx.x * 64 + wave * 16;
  const bool act = (row0 < NN);
  if (act && lane < 16) {             // per-node softmax over 7 relations
    int node = row0 + lane;
    float sr[RR], mx = -3.4e38f;
#pragma unroll
    for (int r = 0; r < RR; ++r) { sr[r] = scores[(size_t)r * NN + node]; mx = fmaxf(mx, sr[r]); }
    float sum = 0.f;
#pragma unroll
    for (int r = 0; r < RR; ++r) { sr[r] = __expf(sr[r] - mx); sum += sr[r]; }
    float inv = 1.f / sum;
#pragma unroll
    for (int r = 0; r < RR; ++r) att[(wave * RR + r) * 16 + lane] = sr[r] * inv;
  }
  __syncthreads();
  if (!act) return;
  float sg = 1.f / (1.f + __expf(-skip[0]));
  v8f z = {};
  v8f acc[8];
#pragma unroll
  for (int nt = 0; nt < 8; ++nt) acc[nt] = z;
  for (int r = 0; r < RR; ++r) {
    const float* rb = rs + ((size_t)r * NN + row0 + nl) * DD;
    v16bf a[4];
#pragma unroll
    for (int ks = 0; ks < 4; ++ks) a[ks] = ld_a_f32_lrelu(rb, ks * 32, h);
    float attr[8];
#pragma unroll
    for (int j = 0; j < 8; ++j) attr[j] = att[(wave * RR + r) * 16 + 8 * h + j];
#pragma unroll
    for (int nt = 0; nt < 8; ++nt) {
      v8f c = {};
#pragma unroll
      for (int ks = 0; ks < 4; ++ks) c = wmma_bf(a[ks], ld_b(WT, DD, nt * 16 + nl, ks * 32, h), c);
      float bvv = bv[nt * 16 + nl];
#pragma unroll
      for (int j = 0; j < 8; ++j) acc[nt][j] += attr[j] * (c[j] + bvv);
    }
  }
#pragma unroll
  for (int nt = 0; nt < 8; ++nt)
#pragma unroll
    for (int j = 0; j < 8; ++j) {
      size_t o = (size_t)(row0 + 8 * h + j) * DD + nt * 16 + nl;
      float x = emb[o];
      float g = 0.5f * x * (1.f + erff(x * 0.70710678118654752f)); // exact GELU
      out[o] = acc[nt][j] + sg * g;
    }
}

// ---------------- host launcher ----------------
extern "C" void kernel_launch(void* const* d_in, const int* in_sizes, int n_in,
                              void* d_out, int out_size, void* d_ws, size_t ws_size,
                              hipStream_t stream) {
  (void)in_sizes; (void)n_in; (void)out_size; (void)ws_size;
  const float* company    = (const float*)d_in[0];
  const float* person     = (const float*)d_in[1];
  const float* edge_w     = (const float*)d_in[2];
  const float* W_com      = (const float*)d_in[3];
  const float* W_per      = (const float*)d_in[4];
  const float* Wq         = (const float*)d_in[5];
  const float* bq         = (const float*)d_in[6];
  const float* Wk         = (const float*)d_in[7];
  const float* bk         = (const float*)d_in[8];
  const float* Wv         = (const float*)d_in[9];
  const float* bv         = (const float*)d_in[10];
  const float* pri        = (const float*)d_in[11];
  const float* W_rel_pair = (const float*)d_in[12];
  const float* W_rel3     = (const float*)d_in[13];
  const float* skip       = (const float*)d_in[14];
  const int*   edge_index = (const int*)d_in[15];
  // d_in[16] = edge_type: relation blocks are contiguous (ER per relation), unused

  char* ws = (char*)d_ws;
  size_t p = 0;
  auto take = [&](size_t bytes) -> void* {
    void* q = ws + p;
    p += (bytes + 255) & ~(size_t)255;
    return q;
  };
  float*    emb    = (float*)take((size_t)NN * DD * 4);
  __bf16*   embh   = (__bf16*)take((size_t)NN * DD * 2);
  float*    rsbuf  = (float*)take((size_t)RR * NN * DD * 4);
  float*    temp   = (float*)take((size_t)EPR * DD * 4);
  unsigned* menc   = (unsigned*)take((size_t)NN * DD * 4);
  float*    sden   = (float*)take((size_t)NN * DD * 4);
  unsigned* m3     = (unsigned*)take((size_t)NN * 4);
  float*    s3     = (float*)take((size_t)NN * 4);
  float*    scores = (float*)take((size_t)RR * NN * 4);

  const int ntiles = (NN + 63) / 64;
  const int etiles = (EPR + 63) / 64;
  const int eelem  = (EPR * DD) / 256;

  k_zero<<<8192, 256, 0, stream>>>(rsbuf, (long long)RR * NN * DD);
  k_embed<<<ntiles, 128, 0, stream>>>(company, person, W_com, W_per, emb, embh);

  for (int i = 0; i < RR; ++i) {
    if (i == 3) continue;
    int e0 = i * EPR;
    k_init_ms<<<8192, 256, 0, stream>>>(menc, sden, (long long)NN * DD);
    k_edge_attn<<<etiles, 128, 0, stream>>>(embh, edge_index,
        W_rel_pair + (size_t)i * 2 * DD * DD, temp, menc, e0);
    k_exp_sum<<<eelem, 256, 0, stream>>>(edge_index, temp, menc, sden, e0);
    k_msg<<<eelem, 256, 0, stream>>>(edge_index, temp, sden, emb,
        rsbuf + (size_t)i * NN * DD, e0);
  }

  // relation 3: scalar-weight softmax branch
  k_init_ms<<<4096, 256, 0, stream>>>(m3, s3, (long long)NN);
  k_w_max3<<<(EPR + 255) / 256, 256, 0, stream>>>(edge_index, edge_w, m3, 3 * EPR);
  k_w_sum3<<<(EPR + 255) / 256, 256, 0, stream>>>(edge_index, edge_w, m3, s3, 3 * EPR);
  k_edge3<<<etiles, 128, 0, stream>>>(embh, edge_index, W_rel3, edge_w, m3, s3,
      rsbuf + (size_t)3 * NN * DD, 3 * EPR);

  dim3 gs(ntiles, RR);
  k_scores<<<gs, 128, 0, stream>>>(embh, rsbuf, Wq, bq, Wk, bk, pri, scores);
  k_out<<<ntiles, 128, 0, stream>>>(rsbuf, Wv, bv, scores, emb, skip, (float*)d_out);
}
